// Qwen3MoEBlock_74440373175028
// MI455X (gfx1250) — matrix operand order repaired and checked by reading, not compiled, not run on hardware
//
#include <hip/hip_runtime.h>
#include <hip/hip_bf16.h>

// ---------------------------------------------------------------------------
// Qwen3-style MoE block for MI455X (gfx1250), sparse top-4 formulation.
//   T=8192 tokens, H=1024, I=512, E=32 experts, K=4.
// Strategy:
//   * One-time preprocessing: x -> bf16; weights -> bf16, TRANSPOSED so B
//     fragments are contiguous 16B/32B runs (global_load_b128, no per-element
//     conversion VALU in the GEMM hot loop).
//   * Router in exact f32 (matches reference softmax/top-k ordering).
//   * Expert GEMMs with v_wmma_f32_16x16x32_bf16, f32 accumulation.
//   * Weighted scatter-add via global_atomic_add_f32 (unsafeAtomicAdd).
// ---------------------------------------------------------------------------

typedef __attribute__((ext_vector_type(16))) __bf16 v16bf;
typedef __attribute__((ext_vector_type(8)))  __bf16 v8bf;
typedef __attribute__((ext_vector_type(4)))  __bf16 v4bf;
typedef __attribute__((ext_vector_type(8)))  float  v8f;

#define T_TOK 8192
#define HDIM  1024
#define IDIM  512
#define NEXP  32
#define TOPK  4
#define MT    32            // tokens per expert-tile (block)
#define HSTR  (IDIM + 8)    // LDS row stride for h tile (520*2B rows, 16B aligned)

__device__ __forceinline__ __bf16 f2bf(float f) {
    unsigned u = __float_as_uint(f);
    u += 0x7FFFu + ((u >> 16) & 1u);          // round-to-nearest-even
    unsigned short hs = (unsigned short)(u >> 16);
    return __builtin_bit_cast(__bf16, hs);
}

__device__ __forceinline__ v8f wmma_bf16(v16bf a, v16bf b, v8f c) {
    // (neg_a, A, neg_b, B, c_mod, C, reuse_a, reuse_b)
    return __builtin_amdgcn_wmma_f32_16x16x32_bf16(false, a, false, b,
                                                   (short)0, c, false, false);
}

__device__ __forceinline__ v16bf cat8(v8bf lo, v8bf hi) {
    return __builtin_shufflevector(lo, hi, 0, 1, 2, 3, 4, 5, 6, 7,
                                   8, 9, 10, 11, 12, 13, 14, 15);
}

// A fragment (16x32 bf16) from a bf16 row (global or LDS).
// Lane L: m = L&15; elems 0..7 -> K = 8*hi + j ; elems 8..15 -> K = 16+8*hi+j
__device__ __forceinline__ v16bf load_a_frag(const __bf16* row, int k0, int hi) {
    v8bf lo = *(const v8bf*)(row + k0 + 8 * hi);
    v8bf hh = *(const v8bf*)(row + k0 + 16 + 8 * hi);
    return cat8(lo, hh);
}

// B fragment (32x16 bf16) from a TRANSPOSED bf16 weight matrix Wt[n][k],
// leading dim ldk. Lane L: n = L&15; elems 0..7 -> K = 8*hi + j ; elems 8..15 -> K = 16+8*hi+j
__device__ __forceinline__ v16bf load_b_frag(const __bf16* __restrict__ Wt,
                                             int col, int ldk, int k0, int hi) {
    const __bf16* p = Wt + (size_t)col * ldk + k0 + 8 * hi;
    v8bf lo = *(const v8bf*)p;
    v8bf hh = *(const v8bf*)(p + 16);
    return cat8(lo, hh);
}

// ---------------------------------------------------------------------------
// Preprocess A: x (f32) -> xb (bf16), same layout. 4 elements per thread.
// ---------------------------------------------------------------------------
__global__ void cvt_x_kernel(const float* __restrict__ in,
                             __bf16* __restrict__ out, int n) {
    int i = (blockIdx.x * blockDim.x + threadIdx.x) * 4;
    if (i >= n) return;
    float4 f = *(const float4*)(in + i);
    v4bf o;
    o[0] = f2bf(f.x); o[1] = f2bf(f.y); o[2] = f2bf(f.z); o[3] = f2bf(f.w);
    *(v4bf*)(out + i) = o;
}

// ---------------------------------------------------------------------------
// Preprocess B: in[e][R][C] f32  ->  out[e][C][R] bf16 (convert + transpose).
// Grid (C/32, R/32, E), block (32, 8). LDS tile transpose, coalesced I/O.
// ---------------------------------------------------------------------------
__global__ void transpose_cvt_kernel(const float* __restrict__ in,
                                     __bf16* __restrict__ out,
                                     int R, int C) {
    __shared__ float tile[32][33];
    int e = blockIdx.z;
    const float* inp = in + (size_t)e * R * C;
    __bf16* outp = out + (size_t)e * R * C;
    int c0 = blockIdx.x * 32, r0 = blockIdx.y * 32;
#pragma unroll
    for (int j = 0; j < 4; ++j) {
        int r = threadIdx.y * 4 + j;
        tile[r][threadIdx.x] = inp[(size_t)(r0 + r) * C + c0 + threadIdx.x];
    }
    __syncthreads();
#pragma unroll
    for (int j = 0; j < 4; ++j) {
        int c = threadIdx.y * 4 + j;
        outp[(size_t)(c0 + c) * R + r0 + threadIdx.x] = f2bf(tile[threadIdx.x][c]);
    }
}

// ---------------------------------------------------------------------------
// Kernel: router — one wave32 per token. Exact f32 softmax + top-4.
// ---------------------------------------------------------------------------
__global__ void router_kernel(const float* __restrict__ x,
                              const float* __restrict__ gate_w,
                              int* __restrict__ top_idx,
                              float* __restrict__ top_w,
                              int* __restrict__ counts) {
    int wave = threadIdx.x >> 5;
    int lane = threadIdx.x & 31;
    int t = blockIdx.x * (blockDim.x >> 5) + wave;
    if (t >= T_TOK) return;
    const float* xrow = x + (size_t)t * HDIM;

    float mylogit = 0.f;
    for (int e = 0; e < NEXP; ++e) {
        const float* g = gate_w + (size_t)e * HDIM;
        float s = 0.f;
        for (int h = lane; h < HDIM; h += 32) s += xrow[h] * g[h];
#pragma unroll
        for (int off = 16; off > 0; off >>= 1) s += __shfl_xor(s, off, 32);
        if (lane == e) mylogit = s;
    }

    float m = mylogit;
#pragma unroll
    for (int off = 16; off > 0; off >>= 1) m = fmaxf(m, __shfl_xor(m, off, 32));
    float p = __expf(mylogit - m);
    float s = p;
#pragma unroll
    for (int off = 16; off > 0; off >>= 1) s += __shfl_xor(s, off, 32);
    p /= s;

    float pv = p;
    float wk[4]; int ik[4];
#pragma unroll
    for (int k = 0; k < TOPK; ++k) {
        float v = pv; int idx = lane;
#pragma unroll
        for (int off = 16; off > 0; off >>= 1) {
            float ov = __shfl_xor(v, off, 32);
            int   oi = __shfl_xor(idx, off, 32);
            if (ov > v || (ov == v && oi < idx)) { v = ov; idx = oi; }
        }
        wk[k] = v; ik[k] = idx;
        if (lane == idx) pv = -1.f;
    }
    float wsum = wk[0] + wk[1] + wk[2] + wk[3];
    if (lane < TOPK) {
        float wv = (lane == 0) ? wk[0] : (lane == 1) ? wk[1]
                 : (lane == 2) ? wk[2] : wk[3];
        int iv   = (lane == 0) ? ik[0] : (lane == 1) ? ik[1]
                 : (lane == 2) ? ik[2] : ik[3];
        top_w[t * TOPK + lane]   = wv / wsum;
        top_idx[t * TOPK + lane] = iv;
        atomicAdd(&counts[iv], 1);
    }
}

__global__ void prefix_kernel(const int* __restrict__ counts,
                              int* __restrict__ offsets) {
    if (threadIdx.x == 0 && blockIdx.x == 0) {
        int acc = 0;
        for (int e = 0; e < NEXP; ++e) { offsets[e] = acc; acc += counts[e]; }
        offsets[NEXP] = acc;
    }
}

__global__ void scatter_kernel(const int* __restrict__ top_idx,
                               const float* __restrict__ top_w,
                               const int* __restrict__ offsets,
                               int* __restrict__ fill,
                               int* __restrict__ bucket_tok,
                               float* __restrict__ bucket_w) {
    int i = blockIdx.x * blockDim.x + threadIdx.x;
    if (i >= T_TOK * TOPK) return;
    int e = top_idx[i];
    int pos = offsets[e] + atomicAdd(&fill[e], 1);
    bucket_tok[pos] = i >> 2;
    bucket_w[pos]   = top_w[i];
}

// ---------------------------------------------------------------------------
// Expert SwiGLU MLP for a 32-token tile of one expert (all-bf16 operands).
//   Phase 1: h = silu(X Wg) * (X Wu) * combine_w  -> LDS (bf16, 32x512)
//   Phase 2: y += h Wd                            -> atomic f32 scatter-add
// ---------------------------------------------------------------------------
__launch_bounds__(256, 1)
__global__ void expert_kernel(const __bf16* __restrict__ xb,
                              const __bf16* __restrict__ wgT,  // [E][I][H]
                              const __bf16* __restrict__ wuT,  // [E][I][H]
                              const __bf16* __restrict__ wdT,  // [E][H][I]
                              const int* __restrict__ offsets,
                              const int* __restrict__ bucket_tok,
                              const float* __restrict__ bucket_w,
                              float* __restrict__ out) {
    __shared__ __bf16 h_lds[MT][HSTR];
    __shared__ float  cw_lds[MT];
    __shared__ int    tok_lds[MT];

    int e   = blockIdx.x;
    int beg = offsets[e];
    int ne  = offsets[e + 1] - beg;
    int m0  = blockIdx.y * MT;
    if (m0 >= ne) return;

    int wave = threadIdx.x >> 5;
    int lane = threadIdx.x & 31;
    int hi   = lane >> 4;
    int ln   = lane & 15;

    if (threadIdx.x < MT) {
        int rr = m0 + threadIdx.x;
        int cl = (rr < ne) ? rr : (ne - 1);
        tok_lds[threadIdx.x] = bucket_tok[beg + cl];
        cw_lds[threadIdx.x]  = (rr < ne) ? bucket_w[beg + rr] : 0.f;
    }
    __syncthreads();

    const __bf16* Wg = wgT + (size_t)e * HDIM * IDIM;
    const __bf16* Wu = wuT + (size_t)e * HDIM * IDIM;
    const __bf16* Wd = wdT + (size_t)e * IDIM * HDIM;

    // ---------------- Phase 1: gate/up GEMM over K = H ----------------
    const __bf16* arow0 = xb + (size_t)tok_lds[ln]      * HDIM;
    const __bf16* arow1 = xb + (size_t)tok_lds[16 + ln] * HDIM;
    int colbase = wave * 64;                 // 8 waves * 64 cols = I

    v8f accg[2][4], accu[2][4];
#pragma unroll
    for (int ms = 0; ms < 2; ++ms)
#pragma unroll
        for (int nt = 0; nt < 4; ++nt) { accg[ms][nt] = {}; accu[ms][nt] = {}; }

    for (int k0 = 0; k0 < HDIM; k0 += 32) {
        v16bf a0 = load_a_frag(arow0, k0, hi);
        v16bf a1 = load_a_frag(arow1, k0, hi);
#pragma unroll
        for (int nt = 0; nt < 4; ++nt) {
            int col = colbase + nt * 16 + ln;
            v16bf bg = load_b_frag(Wg, col, HDIM, k0, hi);
            v16bf bu = load_b_frag(Wu, col, HDIM, k0, hi);
            accg[0][nt] = wmma_bf16(a0, bg, accg[0][nt]);
            accg[1][nt] = wmma_bf16(a1, bg, accg[1][nt]);
            accu[0][nt] = wmma_bf16(a0, bu, accu[0][nt]);
            accu[1][nt] = wmma_bf16(a1, bu, accu[1][nt]);
        }
    }

    // SwiGLU + combine-weight scale; write h tile to LDS.
    // C layout: vgpr r -> m = r + 8*hi, n = ln.
#pragma unroll
    for (int ms = 0; ms < 2; ++ms)
#pragma unroll
        for (int nt = 0; nt < 4; ++nt)
#pragma unroll
            for (int r = 0; r < 8; ++r) {
                int row = ms * 16 + 8 * hi + r;
                float g = accg[ms][nt][r];
                float u = accu[ms][nt][r];
                float sig = 1.f / (1.f + __expf(-g));
                h_lds[row][colbase + nt * 16 + ln] =
                    f2bf(g * sig * u * cw_lds[row]);
            }
    __syncthreads();

    // ---------------- Phase 2: down GEMM over K = I ----------------
    int ob = wave * 128;                     // 8 waves * 128 cols = H
    for (int nt = 0; nt < 8; ++nt) {
        v8f acc0 = {}, acc1 = {};
        int col = ob + nt * 16 + ln;
        for (int k0 = 0; k0 < IDIM; k0 += 32) {
            v16bf b  = load_b_frag(Wd, col, IDIM, k0, hi);
            v16bf a0 = load_a_frag(&h_lds[ln][0],      k0, hi);
            v16bf a1 = load_a_frag(&h_lds[16 + ln][0], k0, hi);
            acc0 = wmma_bf16(a0, b, acc0);
            acc1 = wmma_bf16(a1, b, acc1);
        }
#pragma unroll
        for (int r = 0; r < 8; ++r) {
            int row0 = 8 * hi + r;
            int row1 = 16 + 8 * hi + r;
            if (m0 + row0 < ne)
                unsafeAtomicAdd(&out[(size_t)tok_lds[row0] * HDIM + col], acc0[r]);
            if (m0 + row1 < ne)
                unsafeAtomicAdd(&out[(size_t)tok_lds[row1] * HDIM + col], acc1[r]);
        }
    }
}

// ---------------------------------------------------------------------------
// Host launcher (graph-capture safe: only stream-ordered ops).
// ---------------------------------------------------------------------------
extern "C" void kernel_launch(void* const* d_in, const int* in_sizes, int n_in,
                              void* d_out, int out_size, void* d_ws, size_t ws_size,
                              hipStream_t stream) {
    (void)in_sizes; (void)n_in; (void)out_size; (void)ws_size;
    const float* x      = (const float*)d_in[0];
    const float* gate_w = (const float*)d_in[1];
    const float* w_gate = (const float*)d_in[2];
    const float* w_up   = (const float*)d_in[3];
    const float* w_down = (const float*)d_in[4];
    float* out = (float*)d_out;

    const size_t TK = (size_t)T_TOK * TOPK;            // 32768
    const size_t WB = (size_t)NEXP * HDIM * IDIM;      // 16.8M elems per weight
    char* ws = (char*)d_ws;
    // --- routing block (first 1 MB) ---
    int*    top_idx = (int*)(ws + 0);                  // TK ints
    float*  top_w   = (float*)(ws + TK * 4);
    int*    counts  = (int*)(ws + TK * 8);
    int*    offsets = (int*)(ws + TK * 8 + 256);
    int*    fill    = (int*)(ws + TK * 8 + 768);
    int*    btok    = (int*)(ws + TK * 8 + 1024);
    float*  bw      = (float*)(ws + TK * 8 + 1024 + TK * 4);
    // --- bf16 staging ---
    size_t off = 1u << 20;
    __bf16* xb  = (__bf16*)(ws + off); off += (size_t)T_TOK * HDIM * 2;  // 16 MB
    __bf16* wgT = (__bf16*)(ws + off); off += WB * 2;                    // 32 MB
    __bf16* wuT = (__bf16*)(ws + off); off += WB * 2;                    // 32 MB
    __bf16* wdT = (__bf16*)(ws + off);                                   // 32 MB

    hipMemsetAsync(out, 0, (size_t)T_TOK * HDIM * sizeof(float), stream);
    hipMemsetAsync(counts, 0, 1024, stream);  // counts + offsets + fill

    // Preprocessing: bf16 conversion (+ transpose for weights), once.
    cvt_x_kernel<<<(T_TOK * HDIM / 4 + 255) / 256, 256, 0, stream>>>(
        x, xb, T_TOK * HDIM);
    dim3 tb(32, 8);
    transpose_cvt_kernel<<<dim3(IDIM / 32, HDIM / 32, NEXP), tb, 0, stream>>>(
        w_gate, wgT, HDIM, IDIM);   // [E][H][I] -> [E][I][H]
    transpose_cvt_kernel<<<dim3(IDIM / 32, HDIM / 32, NEXP), tb, 0, stream>>>(
        w_up, wuT, HDIM, IDIM);
    transpose_cvt_kernel<<<dim3(HDIM / 32, IDIM / 32, NEXP), tb, 0, stream>>>(
        w_down, wdT, IDIM, HDIM);   // [E][I][H] -> [E][H][I]

    // Routing.
    router_kernel<<<T_TOK / 8, 256, 0, stream>>>(x, gate_w, top_idx, top_w, counts);
    prefix_kernel<<<1, 32, 0, stream>>>(counts, offsets);
    scatter_kernel<<<(T_TOK * TOPK + 255) / 256, 256, 0, stream>>>(
        top_idx, top_w, offsets, fill, btok, bw);

    // Expert GEMMs.
    dim3 grid(NEXP, T_TOK / MT);   // worst case: all tokens on one expert
    expert_kernel<<<grid, 256, 0, stream>>>(xb, wgT, wuT, wdT,
                                            offsets, btok, bw, out);
}
